// CAT_Attention_74242804678959
// MI455X (gfx1250) — matrix operand order repaired and checked by reading, not compiled, not run on hardware
//
#include <hip/hip_runtime.h>
#include <hip/hip_bf16.h>

// ---------------------------------------------------------------------------
// CAT attention for gfx1250 (MI455X), bf16 WMMA pipeline + TDM A-panel staging.
// B=1, BLOCK=2048, CHUNK=64, DIM=1024, CAT_DIM=1024, NH=16, HD=64
// NC=32, W=65, S=2081 (padded to SP=2096 = 131*16)
// ---------------------------------------------------------------------------

typedef unsigned short ushort_t;
typedef __attribute__((ext_vector_type(16))) __bf16 v16bf;
typedef __attribute__((ext_vector_type(8)))  float  v8f;
typedef __attribute__((ext_vector_type(4)))  unsigned int u32x4;
typedef __attribute__((ext_vector_type(8)))  int i32x8;
typedef __attribute__((ext_vector_type(4)))  int i32x4;

#if __has_builtin(__builtin_amdgcn_tensor_load_to_lds)
#define HAVE_TDM 1
#else
#define HAVE_TDM 0
#endif

union V16 { uint4 q2[2]; v16bf v; };

__device__ __forceinline__ ushort_t f2bf(float f) {
    union { float f; unsigned u; } x; x.f = f;
    unsigned r = x.u + 0x7FFFu + ((x.u >> 16) & 1u);   // round-to-nearest-even
    return (ushort_t)(r >> 16);
}

// ---------------------------------------------------------------------------
// elementwise cast f32 -> bf16
// ---------------------------------------------------------------------------
__global__ void cast_bf16_kernel(const float* __restrict__ src,
                                 ushort_t* __restrict__ dst, int n) {
    for (int i = blockIdx.x * blockDim.x + threadIdx.x; i < n;
         i += gridDim.x * blockDim.x)
        dst[i] = f2bf(src[i]);
}

// ---------------------------------------------------------------------------
// cast + transpose: src[K][N] f32 -> dst[N][K] bf16  (32x32 LDS tiles)
// ---------------------------------------------------------------------------
__global__ __launch_bounds__(256)
void transpose_cast_kernel(const float* __restrict__ src,
                           ushort_t* __restrict__ dst, int K, int N) {
    __shared__ float tile[32][33];
    int n0 = blockIdx.x * 32, k0 = blockIdx.y * 32;
    int tx = threadIdx.x & 31, ty = threadIdx.x >> 5;     // ty: 0..7
    for (int r = ty; r < 32; r += 8)
        tile[r][tx] = src[(size_t)(k0 + r) * N + (n0 + tx)];
    __syncthreads();
    for (int r = ty; r < 32; r += 8)
        dst[(size_t)(n0 + r) * K + (k0 + tx)] = f2bf(tile[tx][r]);
}

// ---------------------------------------------------------------------------
// init h: zero everything (SP*1024 rows incl pad), row 0 <- dummy_fx
// ---------------------------------------------------------------------------
__global__ void init_h_kernel(ushort_t* __restrict__ h,
                              const float* __restrict__ dummy, int total) {
    for (int i = blockIdx.x * blockDim.x + threadIdx.x; i < total;
         i += gridDim.x * blockDim.x)
        h[i] = (i < 1024) ? f2bf(dummy[i]) : (ushort_t)0;
}

// ---------------------------------------------------------------------------
// Row maps for fused scatter/gather GEMMs
// ---------------------------------------------------------------------------
__device__ __forceinline__ int map_xe_row(int m) {            // xe -> h
    return (m >> 6) * 65 + 1 + (m & 63);
}
__device__ __forceinline__ int map_fx_row(int m) {            // fx -> h
    return (m < 31) ? (m + 1) * 65 : 2080;
}
__device__ __forceinline__ int map_final_row(int r) {         // out-gather <- yo
    if (r < 63) return r + 1;
    if (r >= 2047) return 2080;
    int q = r - 63;
    return (1 + q / 64) * 65 + (q % 64);
}

// ---------------------------------------------------------------------------
// TDM: 2-D tile load (16 rows x 64 bf16) global -> LDS.
// D# per CDNA5 ISA ch.8: group0 = {count, lds_addr, global_addr, type=2},
// group1 = {data_size=2B, tensor_dim0=lda, tensor_dim1=big, tile=64x16,
//           tensor_dim0_stride=lda}. Remaining groups zero (2-D tensor).
// ---------------------------------------------------------------------------
#if HAVE_TDM
__device__ __forceinline__ void tdm_load_tile16x64(const ushort_t* gptr,
                                                   unsigned lds_off,
                                                   unsigned lda_elems) {
    unsigned long long ga = (unsigned long long)(uintptr_t)gptr;
    u32x4 g0;
    g0[0] = 1u;                                        // count=1, user mode
    g0[1] = lds_off;                                   // lds_addr
    g0[2] = (unsigned)(ga & 0xFFFFFFFFu);              // global_addr[31:0]
    g0[3] = (unsigned)((ga >> 32) & 0x01FFFFFFu)       // global_addr[56:32]
          | (2u << 30);                                // type = 2 (image)
    i32x8 g1;
    g1[0] = (int)(1u << 16);                           // data_size=1 -> 2 bytes
    g1[1] = (int)((lda_elems & 0xFFFFu) << 16);        // tensor_dim0[15:0]
    g1[2] = (int)((lda_elems >> 16) & 0xFFFFu);        // tensor_dim0[31:16], dim1 lo16=0
    g1[3] = (int)(0x10u | (64u << 16));                // tensor_dim1=1<<20, tile_dim0=64
    g1[4] = (int)16u;                                  // tile_dim1=16, tile_dim2=0
    g1[5] = (int)lda_elems;                            // tensor_dim0_stride[31:0]
    g1[6] = 0;                                         // stride hi, dim1_stride lo
    g1[7] = 0;
    i32x4 gz4; gz4[0] = 0; gz4[1] = 0; gz4[2] = 0; gz4[3] = 0;
    i32x8 gz8;
    #pragma unroll
    for (int i = 0; i < 8; ++i) gz8[i] = 0;
    __builtin_amdgcn_tensor_load_to_lds(g0, g1, gz4, gz4, gz8, 0);
}
#endif

// ---------------------------------------------------------------------------
// Generic bf16 WMMA GEMM:
//   C[M][Ncols] = A[M][K] (bf16 row-major, optional gathered rows)
//               * Bt[N][K] (bf16, i.e. B transposed)  + bias
// Each wave computes a 16x64 tile; K-loop step 64 (2 WMMA substeps).
// A panel staged through LDS (double-buffered TDM when use_tdm).
// grid = (M/16, N/512), block = 256 (8 waves).
// amode: 0=identity, 1=final-gather.   cmode: 0=identity, 1=xe->h, 2=fx->h.
// ---------------------------------------------------------------------------
__global__ __launch_bounds__(256)
void gemm_bf16_wmma(const ushort_t* __restrict__ A, int lda,
                    const ushort_t* __restrict__ Bt, int ldb,
                    float* __restrict__ Cf, ushort_t* __restrict__ Cb, int ldc,
                    const float* __restrict__ bias,
                    int K, int amode, int cmode, int use_tdm) {
    __shared__ __align__(16) ushort_t Atile[2][16 * 64];

    const int tid = threadIdx.x;
    const int lane = tid & 31;
    const int wave = tid >> 5;
    const int mt = blockIdx.x;
    const int n0 = (blockIdx.y * 8 + wave) * 64;
    const int mh = lane >> 4;        // K-half for A/B frags
    const int nn = lane & 15;        // row (A) / col (B,C) within tile

    const int am = mt * 16 + nn;
    const int arow = (amode == 1) ? map_final_row(am) : am;
    const ushort_t* Ap = A + (size_t)arow * lda;
    const ushort_t* Bp0 = Bt + (size_t)(n0 + nn) * ldb;

    v8f acc[4];
    #pragma unroll
    for (int t = 0; t < 4; ++t) {
        float b = bias ? bias[n0 + t * 16 + nn] : 0.0f;
        #pragma unroll
        for (int v = 0; v < 8; ++v) acc[t][v] = b;
    }

    const bool tdm = (use_tdm != 0);
#if HAVE_TDM
    if (tdm && wave == 0)
        tdm_load_tile16x64(A + (size_t)(mt * 16) * lda,
                           (unsigned)(uintptr_t)&Atile[0][0], (unsigned)lda);
#endif

    int buf = 0;
    for (int kb = 0; kb < K; kb += 64, buf ^= 1) {
        if (tdm) {
#if HAVE_TDM
            if (wave == 0) {
                if (kb + 64 < K) {
                    tdm_load_tile16x64(A + (size_t)(mt * 16) * lda + (kb + 64),
                                       (unsigned)(uintptr_t)&Atile[buf ^ 1][0],
                                       (unsigned)lda);
                    __builtin_amdgcn_s_wait_tensorcnt(1);  // current tile done
                } else {
                    __builtin_amdgcn_s_wait_tensorcnt(0);  // last tile done
                }
            }
            __syncthreads();
#else
            // cooperative staging fallback (no TDM builtin on this toolchain)
            for (int idx = tid; idx < 128; idx += 256) {
                int r = idx >> 3, c = idx & 7;
                *(uint4*)(&Atile[buf][r * 64 + c * 8]) =
                    *(const uint4*)(A + (size_t)(mt * 16 + r) * lda + kb + c * 8);
            }
            __syncthreads();
#endif
        }

        // ---- load all fragments for this 64-wide K step, then 8 WMMAs ----
        V16 a[2], b[2][4];
        if (tdm) {
            const ushort_t* At = &Atile[buf][nn * 64];
            #pragma unroll
            for (int sub = 0; sub < 2; ++sub) {
                a[sub].q2[0] = *(const uint4*)(At + sub * 32 + mh * 8);
                a[sub].q2[1] = *(const uint4*)(At + sub * 32 + 16 + mh * 8);
            }
        } else {
            #pragma unroll
            for (int sub = 0; sub < 2; ++sub) {
                a[sub].q2[0] = *(const uint4*)(Ap + kb + sub * 32 + mh * 8);
                a[sub].q2[1] = *(const uint4*)(Ap + kb + sub * 32 + 16 + mh * 8);
            }
        }
        #pragma unroll
        for (int sub = 0; sub < 2; ++sub) {
            #pragma unroll
            for (int t = 0; t < 4; ++t) {
                const ushort_t* Bp =
                    Bp0 + (size_t)t * 16 * ldb + kb + sub * 32 + mh * 8;
                b[sub][t].q2[0] = *(const uint4*)(Bp);
                b[sub][t].q2[1] = *(const uint4*)(Bp + 16);
            }
        }
        #pragma unroll
        for (int sub = 0; sub < 2; ++sub)
            #pragma unroll
            for (int t = 0; t < 4; ++t)
                acc[t] = __builtin_amdgcn_wmma_f32_16x16x32_bf16(
                    false, a[sub].v, false, b[sub][t].v, (short)0, acc[t],
                    false, false);

        if (tdm) __syncthreads();   // protect LDS buffer reuse
    }

    #pragma unroll
    for (int t = 0; t < 4; ++t) {
        #pragma unroll
        for (int v = 0; v < 8; ++v) {
            int m = mt * 16 + mh * 8 + v;
            int crow = (cmode == 1) ? map_xe_row(m)
                     : (cmode == 2) ? map_fx_row(m) : m;
            int col = n0 + t * 16 + nn;
            float val = acc[t][v];
            if (Cf) Cf[(size_t)crow * ldc + col] = val;
            if (Cb) Cb[(size_t)crow * ldc + col] = f2bf(val);
        }
    }
}

// ---------------------------------------------------------------------------
// RoPE + head split. qkv[SP][3072] f32 -> qh/kh/vh [NH][SP][64] bf16.
// q is pre-scaled by 1/sqrt(HD)=0.125. Rows s>=2081 are zeroed.
// ---------------------------------------------------------------------------
__global__ __launch_bounds__(256)
void rope_split_kernel(const float* __restrict__ qkv,
                       const float* __restrict__ cosb,
                       const float* __restrict__ sinb,
                       ushort_t* __restrict__ qh,
                       ushort_t* __restrict__ kh,
                       ushort_t* __restrict__ vh) {
    const int SP = 2096;
    int s = blockIdx.x;
    for (int col = threadIdx.x; col < 1024; col += 256) {
        int h = col >> 6, d = col & 63;
        ushort_t qv = 0, kv = 0, vv = 0;
        if (s < 2081) {
            float cs = cosb[s * 64 + d], sn = sinb[s * 64 + d];
            const float* row = qkv + (size_t)s * 3072;
            float q1 = row[col];
            float k1 = row[1024 + col];
            float v1 = row[2048 + col];
            int dp = (d < 32) ? d + 32 : d - 32;
            float sgn = (d < 32) ? -1.0f : 1.0f;
            float qr = row[h * 64 + dp];
            float kr = row[1024 + h * 64 + dp];
            qv = f2bf((q1 * cs + sgn * qr * sn) * 0.125f);
            kv = f2bf(k1 * cs + sgn * kr * sn);
            vv = f2bf(v1);
        }
        size_t o = ((size_t)h * SP + s) * 64 + d;
        qh[o] = qv; kh[o] = kv; vh[o] = vv;
    }
}

// ---------------------------------------------------------------------------
// Masked CAT attention. One workgroup per (chunk-block bq, head).
// Key slots (128 = 8 N-tiles): [0,48) summary keys j*65, j<bq (own excluded);
// [48,128) own-block keys t=0..64 (causal t<=p), rest zero-padded.
// Block = 160 threads (5 waves); wave w handles query rows p = w*16..w*16+15.
// ---------------------------------------------------------------------------
__global__ __launch_bounds__(160)
void cat_attn_kernel(const ushort_t* __restrict__ qh,
                     const ushort_t* __restrict__ kh,
                     const ushort_t* __restrict__ vh,
                     ushort_t* __restrict__ ybuf) {
    const int SP = 2096;
    __shared__ __align__(16) ushort_t Ks[128 * 64];     // [key][d]
    __shared__ __align__(16) ushort_t Vt[64 * 128];     // [d][key]
    __shared__ __align__(16) ushort_t Pr[5 * 16 * 128]; // per-wave probs [m][key]

    const int bq = blockIdx.x;      // 0..32
    const int head = blockIdx.y;    // 0..15
    const int tid = threadIdx.x;
    const int lane = tid & 31, wave = tid >> 5;
    const ushort_t* qhg = qh + (size_t)head * SP * 64;
    const ushort_t* khg = kh + (size_t)head * SP * 64;
    const ushort_t* vhg = vh + (size_t)head * SP * 64;

    // ---- stage K rows (gathered) ----
    for (int idx = tid; idx < 128 * 8; idx += 160) {
        int key = idx >> 3, j = idx & 7;
        int s; bool valid;
        if (key < 48) { s = key * 65; valid = (key < bq); }
        else { int t = key - 48; s = bq * 65 + t; valid = (t <= 64) && (s <= 2080); }
        uint4 val = make_uint4(0u, 0u, 0u, 0u);
        if (valid) val = *(const uint4*)(khg + (size_t)s * 64 + j * 8);
        *(uint4*)(&Ks[key * 64 + j * 8]) = val;
    }
    // ---- stage V transposed ----
    if (tid < 128) {
        int key = tid;
        int s; bool valid;
        if (key < 48) { s = key * 65; valid = (key < bq); }
        else { int t = key - 48; s = bq * 65 + t; valid = (t <= 64) && (s <= 2080); }
        uint4 r[8];
        #pragma unroll
        for (int j = 0; j < 8; ++j)
            r[j] = valid ? *(const uint4*)(vhg + (size_t)s * 64 + j * 8)
                         : make_uint4(0u, 0u, 0u, 0u);
        const ushort_t* rp = (const ushort_t*)r;
        #pragma unroll
        for (int d = 0; d < 64; ++d) Vt[d * 128 + key] = rp[d];
    }
    __syncthreads();

    const int mt = wave;
    const int mh = lane >> 4;
    const int nn = lane & 15;

    // ---- Q A-fragments ----
    int qm = mt * 16 + nn;
    int sqa = bq * 65 + qm; if (sqa > 2095) sqa = 2095;  // pad rows are zero
    const ushort_t* Aq = qhg + (size_t)sqa * 64;
    V16 aq[2];
    #pragma unroll
    for (int kk = 0; kk < 2; ++kk) {
        aq[kk].q2[0] = *(const uint4*)(Aq + kk * 32 + mh * 8);
        aq[kk].q2[1] = *(const uint4*)(Aq + kk * 32 + 16 + mh * 8);
    }

    // ---- scores: Q (16x64) @ K^T (64x128) ----
    v8f sc[8];
    #pragma unroll
    for (int nt = 0; nt < 8; ++nt) {
        v8f acc;
        #pragma unroll
        for (int v = 0; v < 8; ++v) acc[v] = 0.0f;
        #pragma unroll
        for (int kk = 0; kk < 2; ++kk) {
            V16 bk;
            const ushort_t* kp = &Ks[(nt * 16 + nn) * 64 + kk * 32 + mh * 8];
            bk.q2[0] = *(const uint4*)(kp);
            bk.q2[1] = *(const uint4*)(kp + 16);
            acc = __builtin_amdgcn_wmma_f32_16x16x32_bf16(
                false, aq[kk].v, false, bk.v, (short)0, acc, false, false);
        }
        sc[nt] = acc;
    }

    // ---- mask + softmax ----
    const float NEG = -1e30f;
    #pragma unroll
    for (int v = 0; v < 8; ++v) {
        int m = mh * 8 + v;
        int p = mt * 16 + m;
        bool qvalid = (p <= 64) && (bq * 65 + p <= 2080);
        float vals[8];
        float mx = NEG;
        #pragma unroll
        for (int nt = 0; nt < 8; ++nt) {
            int i = nt * 16 + nn;
            bool kvalid;
            if (i < 48) kvalid = (i < bq);
            else { int t = i - 48; kvalid = (t <= p) && (t <= 64) && (bq * 65 + t <= 2080); }
            float x = (qvalid && kvalid) ? sc[nt][v] : NEG;
            vals[nt] = x;
            mx = fmaxf(mx, x);
        }
        #pragma unroll
        for (int o = 1; o < 16; o <<= 1) mx = fmaxf(mx, __shfl_xor(mx, o, 32));
        float sum = 0.0f;
        #pragma unroll
        for (int nt = 0; nt < 8; ++nt) {
            float e = (vals[nt] > -1e29f) ? __expf(vals[nt] - mx) : 0.0f;
            vals[nt] = e; sum += e;
        }
        #pragma unroll
        for (int o = 1; o < 16; o <<= 1) sum += __shfl_xor(sum, o, 32);
        float inv = (sum > 0.0f) ? 1.0f / sum : 0.0f;
        #pragma unroll
        for (int nt = 0; nt < 8; ++nt)
            Pr[wave * 2048 + m * 128 + nt * 16 + nn] = f2bf(vals[nt] * inv);
    }

    // ---- y = P (16x128) @ V (128x64) ----
    v8f ya[4];
    #pragma unroll
    for (int t = 0; t < 4; ++t)
        #pragma unroll
        for (int v = 0; v < 8; ++v) ya[t][v] = 0.0f;

    #pragma unroll
    for (int kb = 0; kb < 4; ++kb) {
        V16 ap;
        const ushort_t* pp = &Pr[wave * 2048 + nn * 128 + kb * 32];
        ap.q2[0] = *(const uint4*)(pp + mh * 8);
        ap.q2[1] = *(const uint4*)(pp + 16 + mh * 8);
        #pragma unroll
        for (int dt = 0; dt < 4; ++dt) {
            V16 bv;
            const ushort_t* vp = &Vt[(dt * 16 + nn) * 128 + kb * 32 + mh * 8];
            bv.q2[0] = *(const uint4*)(vp);
            bv.q2[1] = *(const uint4*)(vp + 16);
            ya[dt] = __builtin_amdgcn_wmma_f32_16x16x32_bf16(
                false, ap.v, false, bv.v, (short)0, ya[dt], false, false);
        }
    }

    // ---- store y (merged heads layout [SP][1024]) ----
    #pragma unroll
    for (int dt = 0; dt < 4; ++dt) {
        #pragma unroll
        for (int v = 0; v < 8; ++v) {
            int m = mh * 8 + v;
            int p = mt * 16 + m;
            int s = bq * 65 + p;
            if (p <= 64 && s <= 2080)
                ybuf[(size_t)s * 1024 + head * 64 + dt * 16 + nn] = f2bf(ya[dt][v]);
        }
    }
}

// ---------------------------------------------------------------------------
// Host-side launch
// ---------------------------------------------------------------------------
extern "C" void kernel_launch(void* const* d_in, const int* in_sizes, int n_in,
                              void* d_out, int out_size, void* d_ws, size_t ws_size,
                              hipStream_t stream) {
    (void)in_sizes; (void)n_in; (void)out_size; (void)ws_size;
    const float* x_f      = (const float*)d_in[0];   // 2048x1024
    const float* W_expand = (const float*)d_in[1];   // 1024x1024
    const float* W_comp   = (const float*)d_in[2];   // 65536x1024
    const float* b_comp   = (const float*)d_in[3];   // 1024
    const float* dummy_fx = (const float*)d_in[4];   // 1024
    const float* W_qkv    = (const float*)d_in[5];   // 1024x3072
    const float* W_o      = (const float*)d_in[6];   // 1024x1024
    const float* W_final  = (const float*)d_in[7];   // 1024x1024
    const float* cosb     = (const float*)d_in[8];   // 2081x64
    const float* sinb     = (const float*)d_in[9];   // 2081x64
    float* out = (float*)d_out;                      // 2048x1024

    const int SP = 2096;
    size_t off = 0;
    auto alloc = [&](size_t bytes) -> char* {
        char* p = (char*)d_ws + off;
        off += (bytes + 255) & ~(size_t)255;
        return p;
    };
    ushort_t* xb    = (ushort_t*)alloc((size_t)2048 * 1024 * 2);
    ushort_t* WeT   = (ushort_t*)alloc((size_t)1024 * 1024 * 2);
    ushort_t* WqkvT = (ushort_t*)alloc((size_t)3072 * 1024 * 2);
    ushort_t* WoT   = (ushort_t*)alloc((size_t)1024 * 1024 * 2);
    ushort_t* WfT   = (ushort_t*)alloc((size_t)1024 * 1024 * 2);
    ushort_t* WcT   = (ushort_t*)alloc((size_t)1024 * 65536 * 2);
    ushort_t* h     = (ushort_t*)alloc((size_t)SP * 1024 * 2);
    float*    qkv   = (float*)   alloc((size_t)SP * 3072 * 4);
    ushort_t* qhb   = (ushort_t*)alloc((size_t)16 * SP * 64 * 2);
    ushort_t* khb   = (ushort_t*)alloc((size_t)16 * SP * 64 * 2);
    ushort_t* vhb   = (ushort_t*)alloc((size_t)16 * SP * 64 * 2);
    ushort_t* ybuf  = (ushort_t*)alloc((size_t)SP * 1024 * 2);
    ushort_t* yob   = (ushort_t*)alloc((size_t)SP * 1024 * 2);

    // 1) casts / transposes
    cast_bf16_kernel<<<2048, 256, 0, stream>>>(x_f, xb, 2048 * 1024);
    transpose_cast_kernel<<<dim3(32, 32),   256, 0, stream>>>(W_expand, WeT, 1024, 1024);
    transpose_cast_kernel<<<dim3(96, 32),   256, 0, stream>>>(W_qkv,  WqkvT, 1024, 3072);
    transpose_cast_kernel<<<dim3(32, 32),   256, 0, stream>>>(W_o,      WoT, 1024, 1024);
    transpose_cast_kernel<<<dim3(32, 32),   256, 0, stream>>>(W_final,  WfT, 1024, 1024);
    transpose_cast_kernel<<<dim3(32, 2048), 256, 0, stream>>>(W_comp,   WcT, 65536, 1024);

    // 2) h init (zeros + dummy_fx in row 0)
    init_h_kernel<<<8384, 256, 0, stream>>>(h, dummy_fx, SP * 1024);

    // 3) fx = x.reshape(32,65536) @ W_comp + b_comp  -> scattered into h
    gemm_bf16_wmma<<<dim3(2, 2), 256, 0, stream>>>(
        xb, 65536, WcT, 65536, nullptr, h, 1024, b_comp, 65536, 0, 2, 1);

    // 4) xe = x @ W_expand -> scattered into h
    gemm_bf16_wmma<<<dim3(128, 2), 256, 0, stream>>>(
        xb, 1024, WeT, 1024, nullptr, h, 1024, nullptr, 1024, 0, 1, 1);

    // 5) qkv = h @ W_qkv
    gemm_bf16_wmma<<<dim3(131, 6), 256, 0, stream>>>(
        h, 1024, WqkvT, 1024, qkv, nullptr, 3072, nullptr, 1024, 0, 0, 1);

    // 6) RoPE + head split (+ q scaled by 1/sqrt(64))
    rope_split_kernel<<<SP, 256, 0, stream>>>(qkv, cosb, sinb, qhb, khb, vhb);

    // 7) masked CAT attention
    cat_attn_kernel<<<dim3(33, 16), 160, 0, stream>>>(qhb, khb, vhb, ybuf);

    // 8) yo = y @ W_o
    gemm_bf16_wmma<<<dim3(131, 2), 256, 0, stream>>>(
        ybuf, 1024, WoT, 1024, nullptr, yob, 1024, nullptr, 1024, 0, 0, 1);

    // 9) out = gather(yo) @ W_final (non-contiguous A rows -> direct loads)
    gemm_bf16_wmma<<<dim3(128, 2), 256, 0, stream>>>(
        yob, 1024, WfT, 1024, out, nullptr, 1024, nullptr, 1024, 1, 0, 0);
}
